// FluxAttention_42743514530442
// MI455X (gfx1250) — matrix operand order repaired and checked by reading, not compiled, not run on hardware
//
#include <hip/hip_runtime.h>

typedef __attribute__((ext_vector_type(16))) __bf16 v16bf;
typedef __attribute__((ext_vector_type(8)))  __bf16 v8bf;
typedef __attribute__((ext_vector_type(4)))  __bf16 v4bf;
typedef __attribute__((ext_vector_type(8)))  float  v8f;

#define D_   3072
#define H_   24
#define DH_  128
#define LT_  512
#define LI_  2048
#define S_   2560   // LT_ + LI_

__device__ __forceinline__ v8f vzero8() {
  v8f z;
  #pragma unroll
  for (int e = 0; e < 8; ++e) z[e] = 0.0f;
  return z;
}

__device__ __forceinline__ v16bf combine16(v8bf lo, v8bf hi) {
  v16bf r;
  #pragma unroll
  for (int e = 0; e < 8; ++e) { r[e] = lo[e]; r[8 + e] = hi[e]; }
  return r;
}

__device__ __forceinline__ v8f wmma_bf16(v16bf a, v16bf b, v8f c) {
  // D = A(16x32 bf16) x B(32x16 bf16) + C(16x16 f32)
  return __builtin_amdgcn_wmma_f32_16x16x32_bf16(
      /*neg_a=*/false, a, /*neg_b=*/false, b,
      /*c_mod=*/(short)0, c, /*reuse_a=*/false, /*reuse_b=*/false);
}

// ---------------------------------------------------------------------------
// GEMM: C[M,N] = A[M,K](f32 -> bf16) * W[K,N](f32 -> bf16) + bias[N]
// Block: 128 threads = 4 waves. Block tile 64(M) x 128(N), K-step 64 per
// staging round (2 WMMA k-steps). Wave (wm,wn) computes 32x64 = 2x4 tiles,
// 16 WMMAs per barrier pair.
// ---------------------------------------------------------------------------
__global__ __launch_bounds__(128)
void gemm_bf16_wmma(const float* __restrict__ A, const float* __restrict__ W,
                    const float* __restrict__ bias, float* __restrict__ C,
                    int M, int K, int N) {
  __shared__ __bf16 As[64][72];   // 64 x 64 (+8 pad), rows 144B (16B aligned)
  __shared__ __bf16 Bs[128][72];  // W tile stored transposed: [n][k]

  const int tid  = threadIdx.x;
  const int lane = tid & 31;
  const int wave = tid >> 5;
  const int wm   = wave & 1;
  const int wn   = wave >> 1;
  const int m0   = blockIdx.y * 64;
  const int n0   = blockIdx.x * 128;
  const int l16  = lane & 15;

  v8f acc[2][4];
  #pragma unroll
  for (int i = 0; i < 2; ++i)
    #pragma unroll
    for (int j = 0; j < 4; ++j) acc[i][j] = vzero8();

  for (int k0 = 0; k0 < K; k0 += 64) {
    // Stage A tile 64x64 (f32 -> bf16): 1024 float4 slots, 8 per thread
    #pragma unroll
    for (int it = 0; it < 8; ++it) {
      int idx = tid + 128 * it;
      int row = idx >> 4;             // 16 float4 per row
      int c4  = (idx & 15) * 4;
      const float* ap = A + (size_t)(m0 + row) * K + k0 + c4;
      float4 a4 = *(const float4*)ap;
      if (it == 0 && k0 + 64 < K) __builtin_prefetch(ap + 64, 0, 0);
      v4bf pk;
      pk[0] = (__bf16)a4.x; pk[1] = (__bf16)a4.y;
      pk[2] = (__bf16)a4.z; pk[3] = (__bf16)a4.w;
      *(v4bf*)&As[row][c4] = pk;
    }
    // Stage W tile 64x128 transposed (f32 -> bf16): 2048 float4, 16 per thread
    #pragma unroll
    for (int it = 0; it < 16; ++it) {
      int idx = tid + 128 * it;
      int kk  = idx >> 5;             // 32 float4 per k-row
      int c4  = (idx & 31) * 4;
      const float* wp = W + (size_t)(k0 + kk) * N + n0 + c4;
      float4 w4 = *(const float4*)wp;
      if (it == 0 && k0 + 64 < K) __builtin_prefetch(wp + (size_t)64 * N, 0, 0);
      Bs[c4 + 0][kk] = (__bf16)w4.x;
      Bs[c4 + 1][kk] = (__bf16)w4.y;
      Bs[c4 + 2][kk] = (__bf16)w4.z;
      Bs[c4 + 3][kk] = (__bf16)w4.w;
    }
    __syncthreads();

    // A fragments: lane<16 -> K {0..7,16..23}; lane>=16 -> K {8..15,24..31}
    const int akoff = (lane < 16) ? 0 : 8;
    v16bf af[2][2];                  // [i-subtile][k-step]
    #pragma unroll
    for (int i = 0; i < 2; ++i) {
      int row = 32 * wm + 16 * i + l16;
      #pragma unroll
      for (int ks = 0; ks < 2; ++ks) {
        v8bf lo = *(const v8bf*)&As[row][32 * ks + akoff];
        v8bf hi = *(const v8bf*)&As[row][32 * ks + 16 + akoff];
        af[i][ks] = combine16(lo, hi);
      }
    }
    // B fragments: lane<16 -> K {0..7,16..23}; lane>=16 -> K {8..15,24..31}
    const int bkoff = akoff;
    #pragma unroll
    for (int j = 0; j < 4; ++j) {
      int col = 64 * wn + 16 * j + l16;
      v16bf bf0 = combine16(*(const v8bf*)&Bs[col][bkoff],
                            *(const v8bf*)&Bs[col][bkoff + 16]);
      v16bf bf1 = combine16(*(const v8bf*)&Bs[col][32 + bkoff],
                            *(const v8bf*)&Bs[col][32 + bkoff + 16]);
      #pragma unroll
      for (int i = 0; i < 2; ++i) {
        acc[i][j] = wmma_bf16(af[i][0], bf0, acc[i][j]);
        acc[i][j] = wmma_bf16(af[i][1], bf1, acc[i][j]);
      }
    }
    __syncthreads();
  }

  // Epilogue: C layout -> lane = N within tile, VGPR r = row (r / r+8 by lane half)
  const int rowadd = (lane < 16) ? 0 : 8;
  #pragma unroll
  for (int i = 0; i < 2; ++i)
    #pragma unroll
    for (int j = 0; j < 4; ++j) {
      int n = n0 + 64 * wn + 16 * j + l16;
      float b = bias ? bias[n] : 0.0f;
      #pragma unroll
      for (int r = 0; r < 8; ++r) {
        int m = m0 + 32 * wm + 16 * i + r + rowadd;
        C[(size_t)m * N + n] = acc[i][j][r] + b;
      }
    }
}

// ---------------------------------------------------------------------------
// Per-(token, head) RMSNorm + RoPE, concat [text, image], write bf16 (H,S,DH).
// One wave per block; lane handles 4 consecutive dh.
// ---------------------------------------------------------------------------
__global__ __launch_bounds__(32)
void qk_norm_rope(const float* __restrict__ src_txt, const float* __restrict__ src_img,
                  const float* __restrict__ sc_txt, const float* __restrict__ sc_img,
                  const float* __restrict__ rcos, const float* __restrict__ rsin,
                  __bf16* __restrict__ dst) {
  const int tok  = blockIdx.x;
  const int h    = blockIdx.y;
  const int lane = threadIdx.x;
  const int d    = lane * 4;

  const float* src; const float* sc; int row;
  if (tok < LT_) { src = src_txt; sc = sc_txt; row = tok; }
  else           { src = src_img; sc = sc_img; row = tok - LT_; }

  float4 x = *(const float4*)(src + (size_t)row * D_ + h * DH_ + d);
  float ss = x.x * x.x + x.y * x.y + x.z * x.z + x.w * x.w;
  #pragma unroll
  for (int o = 16; o >= 1; o >>= 1) ss += __shfl_xor(ss, o, 32);
  float inv = rsqrtf(ss * (1.0f / 128.0f) + 1e-6f);

  float4 s4 = *(const float4*)(sc + d);
  float x0 = x.x * inv * s4.x, x1 = x.y * inv * s4.y;
  float x2 = x.z * inv * s4.z, x3 = x.w * inv * s4.w;

  float4 c4 = *(const float4*)(rcos + (size_t)tok * DH_ + d);
  float4 sn = *(const float4*)(rsin + (size_t)tok * DH_ + d);
  float y0 = x0 * c4.x - x1 * sn.x;
  float y1 = x1 * c4.y + x0 * sn.y;
  float y2 = x2 * c4.z - x3 * sn.z;
  float y3 = x3 * c4.w + x2 * sn.w;

  v4bf o;
  o[0] = (__bf16)y0; o[1] = (__bf16)y1; o[2] = (__bf16)y2; o[3] = (__bf16)y3;
  *(v4bf*)(dst + ((size_t)h * S_ + tok) * DH_ + d) = o;
}

// V -> bf16, layout (H, DH, S) so P x V B-fragments are per-lane contiguous.
__global__ __launch_bounds__(256)
void v_permute(const float* __restrict__ v_txt, const float* __restrict__ v_img,
               __bf16* __restrict__ vt) {
  size_t i = (size_t)blockIdx.x * 256 + threadIdx.x;  // over H_*DH_*S_
  int s  = (int)(i % S_);
  int hd = (int)(i / S_);          // h*DH_ + dh
  int h  = hd / DH_;
  int dh = hd % DH_;
  float v = (s < LT_) ? v_txt[(size_t)s * D_ + h * DH_ + dh]
                      : v_img[(size_t)(s - LT_) * D_ + h * DH_ + dh];
  vt[i] = (__bf16)v;
}

// ---------------------------------------------------------------------------
// Flash attention, bf16 WMMA. Block = 4 waves; each wave owns 16 query rows
// of one head and runs the full online-softmax K loop (Bc = 32).
// Q,K in (H,S,DH) bf16; V in (H,DH,S) bf16; output fp32 (S, H*DH).
// ---------------------------------------------------------------------------
__global__ __launch_bounds__(128)
void flash_attn(const __bf16* __restrict__ Q, const __bf16* __restrict__ Kc,
                const __bf16* __restrict__ Vt, float* __restrict__ O) {
  __shared__ __bf16 Plds[4][16][32];   // per-wave private P patch (no barriers)

  const int tid   = threadIdx.x;
  const int lane  = tid & 31;
  const int w     = tid >> 5;
  const int l16   = lane & 15;
  const int half8 = (lane < 16) ? 0 : 8;   // row offset (C layout) == A-frag K offset
  const int h     = blockIdx.y;
  const int q0    = blockIdx.x * 64 + w * 16;
  const float scale = 0.08838834764831845f;   // 1/sqrt(128)

  const __bf16* qh = Q  + (size_t)h * S_ * DH_;
  const __bf16* kh = Kc + (size_t)h * S_ * DH_;
  const __bf16* vh = Vt + (size_t)h * DH_ * S_;

  // Load Q as 4 A-fragments (dh chunks of 32), kept in registers all loop.
  v16bf qf[4];
  #pragma unroll
  for (int f = 0; f < 4; ++f) {
    const __bf16* p = qh + (size_t)(q0 + l16) * DH_ + 32 * f + half8;
    qf[f] = combine16(*(const v8bf*)p, *(const v8bf*)(p + 16));
  }

  v8f oacc[8];
  #pragma unroll
  for (int t = 0; t < 8; ++t) oacc[t] = vzero8();
  float m[8], l[8];
  #pragma unroll
  for (int r = 0; r < 8; ++r) { m[r] = -1e30f; l[r] = 0.0f; }

  for (int kb = 0; kb < S_; kb += 32) {
    // S = Q x K^T for 2 key sub-tiles of 16
    v8f sa[2];
    #pragma unroll
    for (int nt = 0; nt < 2; ++nt) {
      sa[nt] = vzero8();
      #pragma unroll
      for (int f = 0; f < 4; ++f) {
        const __bf16* p = kh + (size_t)(kb + 16 * nt + l16) * DH_
                             + 32 * f + half8;
        v16bf bk = combine16(*(const v8bf*)p, *(const v8bf*)(p + 16));
        sa[nt] = wmma_bf16(qf[f], bk, sa[nt]);
      }
    }

    // Online softmax. VGPR r holds row (r + half8); N striped over 16-lane group.
    float corr[8];
    #pragma unroll
    for (int r = 0; r < 8; ++r) {
      float t = fmaxf(sa[0][r], sa[1][r]);
      #pragma unroll
      for (int o = 8; o >= 1; o >>= 1) t = fmaxf(t, __shfl_xor(t, o, 32));
      t *= scale;
      float mn = fmaxf(m[r], t);
      corr[r] = __expf(m[r] - mn);
      float p0 = __expf(sa[0][r] * scale - mn);
      float p1 = __expf(sa[1][r] * scale - mn);
      m[r] = mn;
      float ps = p0 + p1;
      #pragma unroll
      for (int o = 8; o >= 1; o >>= 1) ps += __shfl_xor(ps, o, 32);
      l[r] = l[r] * corr[r] + ps;
      // Reshape P (C layout) -> LDS row-major 16x32 patch
      Plds[w][r + half8][l16]      = (__bf16)p0;
      Plds[w][r + half8][16 + l16] = (__bf16)p1;
    }

    // Reload P as an A-matrix fragment (16x32)
    const __bf16* pp = &Plds[w][l16][0];
    v16bf pf = combine16(*(const v8bf*)(pp + half8),
                         *(const v8bf*)(pp + 16 + half8));

    // O = diag(corr)*O + P x V  (8 dh tiles)
    #pragma unroll
    for (int t = 0; t < 8; ++t) {
      #pragma unroll
      for (int r = 0; r < 8; ++r) oacc[t][r] *= corr[r];
      const __bf16* vp = vh + (size_t)(16 * t + l16) * S_ + kb + half8;
      v16bf bv = combine16(*(const v8bf*)vp, *(const v8bf*)(vp + 16));
      oacc[t] = wmma_bf16(pf, bv, oacc[t]);
    }
  }

  // Write normalized output, token-major (S, H*DH) fp32
  #pragma unroll
  for (int t = 0; t < 8; ++t)
    #pragma unroll
    for (int r = 0; r < 8; ++r) {
      int tok = q0 + r + half8;
      O[(size_t)tok * D_ + h * DH_ + 16 * t + l16] = oacc[t][r] / l[r];
    }
}

// ---------------------------------------------------------------------------
extern "C" void kernel_launch(void* const* d_in, const int* in_sizes, int n_in,
                              void* d_out, int out_size, void* d_ws, size_t ws_size,
                              hipStream_t stream) {
  (void)in_sizes; (void)n_in; (void)out_size; (void)ws_size;

  const float* hidden = (const float*)d_in[0];
  const float* enc    = (const float*)d_in[1];
  const float* rcos   = (const float*)d_in[2];
  const float* rsin   = (const float*)d_in[3];
  const float* wq     = (const float*)d_in[4];
  const float* wk     = (const float*)d_in[5];
  const float* wv     = (const float*)d_in[6];
  const float* nq     = (const float*)d_in[7];
  const float* nk     = (const float*)d_in[8];
  const float* naq    = (const float*)d_in[9];
  const float* nak    = (const float*)d_in[10];
  const float* waq    = (const float*)d_in[11];
  const float* baq    = (const float*)d_in[12];
  const float* wak    = (const float*)d_in[13];
  const float* bak    = (const float*)d_in[14];
  const float* wav    = (const float*)d_in[15];
  const float* bav    = (const float*)d_in[16];
  const float* w_out  = (const float*)d_in[17];
  const float* b_out  = (const float*)d_in[18];
  const float* w_add  = (const float*)d_in[19];
  const float* b_add  = (const float*)d_in[20];
  float* out = (float*)d_out;

  char* ws = (char*)d_ws;
  // Region A (projection fp32, dead after v_permute/qk_norm_rope):
  float* qi  = (float*)ws;                          // 2048*3072
  float* ki  = qi  + (size_t)LI_ * D_;
  float* vi  = ki  + (size_t)LI_ * D_;
  float* qt  = vi  + (size_t)LI_ * D_;              // 512*3072
  float* kt  = qt  + (size_t)LT_ * D_;
  float* vtf = kt  + (size_t)LT_ * D_;
  // Region B (bf16 attention operands):
  __bf16* qc = (__bf16*)(vtf + (size_t)LT_ * D_);   // (H,S,DH)
  __bf16* kc = qc + (size_t)H_ * S_ * DH_;          // (H,S,DH)
  __bf16* vT = kc + (size_t)H_ * S_ * DH_;          // (H,DH,S)
  // Region C aliases region A (lifetimes disjoint):
  float* attn = (float*)ws;                          // (S, H*DH) fp32

  const dim3 blk(128);

  // QKV projections (image stream, no bias)
  gemm_bf16_wmma<<<dim3(D_ / 128, LI_ / 64), blk, 0, stream>>>(hidden, wq, nullptr, qi, LI_, D_, D_);
  gemm_bf16_wmma<<<dim3(D_ / 128, LI_ / 64), blk, 0, stream>>>(hidden, wk, nullptr, ki, LI_, D_, D_);
  gemm_bf16_wmma<<<dim3(D_ / 128, LI_ / 64), blk, 0, stream>>>(hidden, wv, nullptr, vi, LI_, D_, D_);
  // QKV projections (text stream, with bias)
  gemm_bf16_wmma<<<dim3(D_ / 128, LT_ / 64), blk, 0, stream>>>(enc, waq, baq, qt, LT_, D_, D_);
  gemm_bf16_wmma<<<dim3(D_ / 128, LT_ / 64), blk, 0, stream>>>(enc, wak, bak, kt, LT_, D_, D_);
  gemm_bf16_wmma<<<dim3(D_ / 128, LT_ / 64), blk, 0, stream>>>(enc, wav, bav, vtf, LT_, D_, D_);

  // RMSNorm + RoPE + concat -> bf16 attention operands
  qk_norm_rope<<<dim3(S_, H_), 32, 0, stream>>>(qt, qi, naq, nq, rcos, rsin, qc);
  qk_norm_rope<<<dim3(S_, H_), 32, 0, stream>>>(kt, ki, nak, nk, rcos, rsin, kc);
  v_permute<<<dim3((H_ * DH_ * S_) / 256), 256, 0, stream>>>(vtf, vi, vT);

  // Attention
  flash_attn<<<dim3(S_ / 64, H_), blk, 0, stream>>>(qc, kc, vT, attn);

  // Output projections: img_out first, then enc_out (reference return order)
  gemm_bf16_wmma<<<dim3(D_ / 128, LI_ / 64), blk, 0, stream>>>(attn + (size_t)LT_ * D_, w_out, b_out, out, LI_, D_, D_);
  gemm_bf16_wmma<<<dim3(D_ / 128, LT_ / 64), blk, 0, stream>>>(attn, w_add, b_add, out + (size_t)LI_ * D_, LT_, D_, D_);
}
